// PatchCore_33054068310141
// MI455X (gfx1250) — matrix operand order repaired and checked by reading, not compiled, not run on hardware
//
#include <hip/hip_runtime.h>
#include <hip/hip_bf16.h>
#include <math.h>

typedef __attribute__((ext_vector_type(16))) __bf16 v16bf;
typedef __attribute__((ext_vector_type(8)))  __bf16 v8bf;
typedef __attribute__((ext_vector_type(8)))  float  v8f;
typedef __attribute__((ext_vector_type(4)))  float  v4f;
typedef __attribute__((ext_vector_type(4)))  unsigned short v4u16;
typedef __attribute__((ext_vector_type(8)))  unsigned short v8u16;

#define N_EMB 12544
#define M_MB  16384
#define DIM   1536
#define BATCH 16
#define PPB   784
#define KC    32
#define LDT   40      // LDS tile row stride (bf16): 80B, 16B aligned, conflict-free frag loads
#define NSPLIT 8

// workspace layout (bytes)
#define OFF_X2     0u
#define OFF_Y2     50176u
#define OFF_MIN64  115712u
#define OFF_MAXP   216064u
#define OFF_SCORE  216128u
#define OFF_NNIDX  216192u
#define OFF_S2D2   216256u
#define OFF_EHI    1264896ull
#define OFF_ELO    39800064ull
#define OFF_MHI    78335232ull
#define OFF_MLO    128666880ull
#define WS_REQ_FAST 178998528ull

// ---- helpers ---------------------------------------------------------------

__device__ __forceinline__ unsigned short bf16_rne(float f) {
    unsigned u = __float_as_uint(f);
    unsigned r = u + 0x7FFFu + ((u >> 16) & 1u);
    return (unsigned short)(r >> 16);
}

__device__ __forceinline__ v16bf mk16(const unsigned short* p0, const unsigned short* p1) {
    v8bf a = *(const v8bf*)p0;
    v8bf b = *(const v8bf*)p1;
    return __builtin_shufflevector(a, b, 0,1,2,3,4,5,6,7,8,9,10,11,12,13,14,15);
}

__device__ __forceinline__ v16bf mk16v(v8bf a, v8bf b) {
    return __builtin_shufflevector(a, b, 0,1,2,3,4,5,6,7,8,9,10,11,12,13,14,15);
}

#define WMMA_BF16(A, B, C) \
    __builtin_amdgcn_wmma_f32_16x16x32_bf16(false, (A), false, (B), (short)0, (C), false, false)

// ---- kernel 1: fused row norms + (optional) bf16 hi/lo conversion ----------

__global__ __launch_bounds__(256)
void prep_kernel(const float* __restrict__ emb, const float* __restrict__ mb,
                 unsigned short* __restrict__ ehi, unsigned short* __restrict__ elo,
                 unsigned short* __restrict__ mhi, unsigned short* __restrict__ mlo,
                 float* __restrict__ x2, float* __restrict__ y2,
                 unsigned long long* __restrict__ min64, int doConvert) {
    int bidx = blockIdx.x;
    const float* src; float* dst; unsigned short* hi; unsigned short* lo; int row;
    if (bidx < N_EMB) { src = emb; dst = x2; hi = ehi; lo = elo; row = bidx; }
    else              { src = mb;  dst = y2; hi = mhi; lo = mlo; row = bidx - N_EMB; }
    const float* p = src + (size_t)row * DIM;
    unsigned short* ph = hi + (size_t)row * DIM;
    unsigned short* pl = lo + (size_t)row * DIM;

    float s = 0.f;
    for (int k = threadIdx.x * 4; k < DIM; k += 1024) {
        v4f v = *(const v4f*)(p + k);
        v4u16 h4, l4;
        #pragma unroll
        for (int j = 0; j < 4; ++j) {
            float f = v[j];
            s += f * f;
            unsigned short hh = bf16_rne(f);
            h4[j] = hh;
            l4[j] = bf16_rne(f - __uint_as_float((unsigned)hh << 16));
        }
        if (doConvert) {
            *(v4u16*)(ph + k) = h4;
            *(v4u16*)(pl + k) = l4;
        }
    }
    __shared__ float red[8];
    for (int m = 16; m >= 1; m >>= 1) s += __shfl_xor(s, m);
    if ((threadIdx.x & 31) == 0) red[threadIdx.x >> 5] = s;
    __syncthreads();
    if (threadIdx.x == 0) {
        float t = 0.f;
        for (int i = 0; i < 8; ++i) t += red[i];
        dst[row] = t;
        if (bidx < N_EMB) min64[row] = 0xFFFFFFFFFFFFFFFFULL;
    }
}

// ---- fast GEMM pipeline pieces ---------------------------------------------

struct Stage {
    v8u16 rh0, rl0, rh1, rl1;                 // B staging (global -> LDS)
    v8bf  a0h0, a0h1, a0l0, a0l1;             // A row-tile 0 fragments
    v8bf  a1h0, a1h1, a1l0, a1l1;             // A row-tile 1 fragments
};

__device__ __forceinline__ void stage_load(Stage& s, int kk,
        const unsigned short* gbh0, const unsigned short* gbl0,
        const unsigned short* gbh1, const unsigned short* gbl1,
        const unsigned short* pah0, const unsigned short* pal0,
        const unsigned short* pah1, const unsigned short* pal1) {
    s.rh0  = *(const v8u16*)(gbh0 + kk);
    s.rl0  = *(const v8u16*)(gbl0 + kk);
    s.rh1  = *(const v8u16*)(gbh1 + kk);
    s.rl1  = *(const v8u16*)(gbl1 + kk);
    s.a0h0 = *(const v8bf*)(pah0 + kk);
    s.a0h1 = *(const v8bf*)(pah0 + kk + 16);
    s.a0l0 = *(const v8bf*)(pal0 + kk);
    s.a0l1 = *(const v8bf*)(pal0 + kk + 16);
    s.a1h0 = *(const v8bf*)(pah1 + kk);
    s.a1h1 = *(const v8bf*)(pah1 + kk + 16);
    s.a1l0 = *(const v8bf*)(pal1 + kk);
    s.a1l1 = *(const v8bf*)(pal1 + kk + 16);
}

__device__ __forceinline__ void stage_store_lds(const Stage& s,
        unsigned short* bh, unsigned short* bl, int soff0, int soff1) {
    *(v8u16*)(bh + soff0) = s.rh0;
    *(v8u16*)(bl + soff0) = s.rl0;
    *(v8u16*)(bh + soff1) = s.rh1;
    *(v8u16*)(bl + soff1) = s.rl1;
}

__device__ __forceinline__ void stage_compute(const Stage& s, v8f (&acc)[2][4],
        const unsigned short* bh, const unsigned short* bl,
        int wc, int lm, int h) {
    v16bf a_hi[2], a_lo[2];
    a_hi[0] = mk16v(s.a0h0, s.a0h1);
    a_lo[0] = mk16v(s.a0l0, s.a0l1);
    a_hi[1] = mk16v(s.a1h0, s.a1h1);
    a_lo[1] = mk16v(s.a1l0, s.a1l1);
    #pragma unroll
    for (int ct = 0; ct < 4; ++ct) {
        const int tg = wc * 4 + ct;
        const unsigned short* pbh = bh + (tg * 16 + lm) * LDT + h * 8;
        const unsigned short* pbl = bl + (tg * 16 + lm) * LDT + h * 8;
        v16bf b_hi = mk16(pbh, pbh + 16);
        v16bf b_lo = mk16(pbl, pbl + 16);
        #pragma unroll
        for (int rt = 0; rt < 2; ++rt) {
            acc[rt][ct] = WMMA_BF16(a_hi[rt], b_hi, acc[rt][ct]);
            acc[rt][ct] = WMMA_BF16(a_hi[rt], b_lo, acc[rt][ct]);
            acc[rt][ct] = WMMA_BF16(a_lo[rt], b_hi, acc[rt][ct]);
        }
    }
}

// ---- kernel 2 (fast): distance GEMM + row argmin from preconverted bf16 ----
// Wave grid 4 row-groups x 2 col-groups (32 rows x 64 cols per wave). B is
// double-buffered in LDS with the K-loop unrolled by 2 (ping-pong register
// sets: no rotation moves, no WMMA->VALU hazard nops). A fragments load
// straight from global memory.

__global__ __launch_bounds__(256)
void dist_argmin_fast(const unsigned short* __restrict__ ehi, const unsigned short* __restrict__ elo,
                      const unsigned short* __restrict__ mhi, const unsigned short* __restrict__ mlo,
                      const float* __restrict__ x2, const float* __restrict__ y2,
                      unsigned long long* __restrict__ min64) {
    __shared__ unsigned short Bh[2][128 * LDT];
    __shared__ unsigned short Bl[2][128 * LDT];

    const int rb   = blockIdx.x;          // 0..97
    const int ns   = blockIdx.y;          // 0..7
    const int rowBase = rb * 128;
    const int tid  = threadIdx.x;
    const int lane = tid & 31;
    const int wv   = tid >> 5;
    const int wg   = wv & 3;              // row group: rows wg*32..wg*32+31
    const int wc   = wv >> 2;             // col group: tiles wc*4..wc*4+3
    const int lm   = lane & 15;
    const int h    = lane >> 4;

    // per-thread B staging slots: 2 chunks of 8 elements
    const int flat0 = tid * 8;
    const int flat1 = flat0 + 2048;
    const int brow0 = flat0 >> 5, bk0 = flat0 & 31;
    const int brow1 = flat1 >> 5, bk1 = flat1 & 31;
    const int soff0 = brow0 * LDT + bk0;
    const int soff1 = brow1 * LDT + bk1;

    // per-lane A row pointers, two row-tiles (rows arow0 and arow0+16)
    const int arow0 = rowBase + wg * 32 + lm;
    const unsigned short* pah0 = ehi + (size_t)arow0 * DIM + h * 8;
    const unsigned short* pal0 = elo + (size_t)arow0 * DIM + h * 8;
    const unsigned short* pah1 = pah0 + (size_t)16 * DIM;
    const unsigned short* pal1 = pal0 + (size_t)16 * DIM;

    float rowmin[2][8];
    int   rowidx[2][8];
    #pragma unroll
    for (int rt = 0; rt < 2; ++rt)
        #pragma unroll
        for (int r = 0; r < 8; ++r) { rowmin[rt][r] = 3.0e38f; rowidx[rt][r] = 0; }

    const v8f vzero = {0.f,0.f,0.f,0.f,0.f,0.f,0.f,0.f};

    for (int nb = 0; nb < 16; ++nb) {
        const int c0 = ns * (M_MB / NSPLIT) + nb * 128;
        const unsigned short* gbh0 = mhi + (size_t)(c0 + brow0) * DIM + bk0;
        const unsigned short* gbl0 = mlo + (size_t)(c0 + brow0) * DIM + bk0;
        const unsigned short* gbh1 = mhi + (size_t)(c0 + brow1) * DIM + bk1;
        const unsigned short* gbl1 = mlo + (size_t)(c0 + brow1) * DIM + bk1;

        v8f acc[2][4];
        #pragma unroll
        for (int rt = 0; rt < 2; ++rt)
            #pragma unroll
            for (int ct = 0; ct < 4; ++ct) acc[rt][ct] = vzero;

        Stage s0, s1;
        stage_load(s0, 0, gbh0, gbl0, gbh1, gbl1, pah0, pal0, pah1, pal1);

        for (int kk = 0; kk < DIM; kk += 2 * KC) {
            // ---- half 0: buffer 0, data set s0 ----
            stage_store_lds(s0, &Bh[0][0], &Bl[0][0], soff0, soff1);
            __syncthreads();
            stage_load(s1, kk + KC, gbh0, gbl0, gbh1, gbl1, pah0, pal0, pah1, pal1);
            stage_compute(s0, acc, &Bh[0][0], &Bl[0][0], wc, lm, h);

            // ---- half 1: buffer 1, data set s1 ----
            stage_store_lds(s1, &Bh[1][0], &Bl[1][0], soff0, soff1);
            __syncthreads();
            if (kk + 2 * KC < DIM)
                stage_load(s0, kk + 2 * KC, gbh0, gbl0, gbh1, gbl1, pah0, pal0, pah1, pal1);
            stage_compute(s1, acc, &Bh[1][0], &Bl[1][0], wc, lm, h);
        }

        // epilogue: d2 = x2 + y2 - 2*dot; running min over cols
        #pragma unroll
        for (int ct = 0; ct < 4; ++ct) {
            int col = c0 + (wc * 4 + ct) * 16 + lm;
            float yc = y2[col];
            #pragma unroll
            for (int rt = 0; rt < 2; ++rt) {
                float xr_base = 0.f; (void)xr_base;
                #pragma unroll
                for (int r = 0; r < 8; ++r) {
                    float xr = x2[rowBase + wg * 32 + rt * 16 + r + 8 * h];
                    float d2 = fmaxf(xr + yc - 2.0f * acc[rt][ct][r], 0.0f);
                    bool better = d2 < rowmin[rt][r];
                    rowidx[rt][r] = better ? col : rowidx[rt][r];
                    rowmin[rt][r] = better ? d2  : rowmin[rt][r];
                }
            }
        }
        // no extra barrier needed: half-1's __syncthreads already orders
        // buffer-0 reuse for the next nb iteration
    }

    // cross-lane reduce within each 16-lane half, then atomic combine
    #pragma unroll
    for (int rt = 0; rt < 2; ++rt) {
        #pragma unroll
        for (int r = 0; r < 8; ++r) {
            float v = rowmin[rt][r];
            int   ix = rowidx[rt][r];
            #pragma unroll
            for (int m = 8; m >= 1; m >>= 1) {
                float ov = __shfl_xor(v, m);
                int   oi = __shfl_xor(ix, m);
                bool take = (ov < v) || (ov == v && oi < ix);
                v  = take ? ov : v;
                ix = take ? oi : ix;
            }
            if (lm == 0) {
                int row = rowBase + wg * 32 + rt * 16 + r + 8 * h;
                unsigned long long key =
                    ((unsigned long long)__float_as_uint(v) << 32) | (unsigned)ix;
                atomicMin(min64 + row, key);
            }
        }
    }
}

// ---- kernel 2 (fallback, ws too small): in-loop conversion version ---------

__global__ __launch_bounds__(256)
void dist_argmin_kernel(const float* __restrict__ emb, const float* __restrict__ mb,
                        const float* __restrict__ x2, const float* __restrict__ y2,
                        unsigned long long* __restrict__ min64) {
    __shared__ unsigned short Ahi[128 * LDT];
    __shared__ unsigned short Alo[128 * LDT];
    __shared__ unsigned short Bhi[128 * LDT];
    __shared__ unsigned short Blo[128 * LDT];

    const int rb   = blockIdx.x;
    const int ns   = blockIdx.y;
    const int rowBase = rb * 128;
    const int tid  = threadIdx.x;
    const int lane = tid & 31;
    const int wv   = tid >> 5;
    const int lm   = lane & 15;
    const int h    = lane >> 4;

    float x2r[8];
    #pragma unroll
    for (int r = 0; r < 8; ++r) x2r[r] = x2[rowBase + wv * 16 + r + 8 * h];

    float rowmin[8];
    int   rowidx[8];
    #pragma unroll
    for (int r = 0; r < 8; ++r) { rowmin[r] = 3.0e38f; rowidx[r] = 0; }

    const v8f vzero = {0.f,0.f,0.f,0.f,0.f,0.f,0.f,0.f};

    for (int nb = 0; nb < 16; ++nb) {
        const int c0 = ns * (M_MB / NSPLIT) + nb * 128;
        v8f acc[8];
        #pragma unroll
        for (int t = 0; t < 8; ++t) acc[t] = vzero;

        for (int kk = 0; kk < DIM; kk += KC) {
            __syncthreads();
            #pragma unroll
            for (int i = 0; i < 4; ++i) {
                int idx  = tid + 256 * i;
                int flat = idx * 4;
                int rowi = flat >> 5;
                int ki   = flat & 31;
                v4f av = *(const v4f*)(emb + (size_t)(rowBase + rowi) * DIM + kk + ki);
                v4f bv = *(const v4f*)(mb  + (size_t)(c0      + rowi) * DIM + kk + ki);
                v4u16 ah, al, bh, bl;
                #pragma unroll
                for (int j = 0; j < 4; ++j) {
                    unsigned short hh = bf16_rne(av[j]);
                    ah[j] = hh;
                    al[j] = bf16_rne(av[j] - __uint_as_float((unsigned)hh << 16));
                    hh = bf16_rne(bv[j]);
                    bh[j] = hh;
                    bl[j] = bf16_rne(bv[j] - __uint_as_float((unsigned)hh << 16));
                }
                int off = rowi * LDT + ki;
                *(v4u16*)(Ahi + off) = ah;
                *(v4u16*)(Alo + off) = al;
                *(v4u16*)(Bhi + off) = bh;
                *(v4u16*)(Blo + off) = bl;
            }
            __syncthreads();

            const unsigned short* pa = Ahi + (wv * 16 + lm) * LDT + h * 8;
            const unsigned short* pl = Alo + (wv * 16 + lm) * LDT + h * 8;
            v16bf a_hi = mk16(pa, pa + 16);
            v16bf a_lo = mk16(pl, pl + 16);
            #pragma unroll
            for (int t = 0; t < 8; ++t) {
                const unsigned short* pbh = Bhi + (t * 16 + lm) * LDT + h * 8;
                const unsigned short* pbl = Blo + (t * 16 + lm) * LDT + h * 8;
                v16bf b_hi = mk16(pbh, pbh + 16);
                v16bf b_lo = mk16(pbl, pbl + 16);
                acc[t] = WMMA_BF16(a_hi, b_hi, acc[t]);
                acc[t] = WMMA_BF16(a_hi, b_lo, acc[t]);
                acc[t] = WMMA_BF16(a_lo, b_hi, acc[t]);
            }
        }

        #pragma unroll
        for (int t = 0; t < 8; ++t) {
            int col = c0 + t * 16 + lm;
            float yc = y2[col];
            #pragma unroll
            for (int r = 0; r < 8; ++r) {
                float d2 = fmaxf(x2r[r] + yc - 2.0f * acc[t][r], 0.0f);
                bool better = d2 < rowmin[r];
                rowidx[r] = better ? col : rowidx[r];
                rowmin[r] = better ? d2  : rowmin[r];
            }
        }
    }

    #pragma unroll
    for (int r = 0; r < 8; ++r) {
        float v = rowmin[r];
        int   ix = rowidx[r];
        #pragma unroll
        for (int m = 8; m >= 1; m >>= 1) {
            float ov = __shfl_xor(v, m);
            int   oi = __shfl_xor(ix, m);
            bool take = (ov < v) || (ov == v && oi < ix);
            v  = take ? ov : v;
            ix = take ? oi : ix;
        }
        if (lm == 0) {
            int row = rowBase + wv * 16 + r + 8 * h;
            unsigned long long key =
                ((unsigned long long)__float_as_uint(v) << 32) | (unsigned)ix;
            atomicMin(min64 + row, key);
        }
    }
}

// ---- kernel 3: per-image argmax of patch scores ----------------------------

__global__ __launch_bounds__(256)
void argmax_kernel(const unsigned long long* __restrict__ min64,
                   int* __restrict__ maxPatch, float* __restrict__ score,
                   int* __restrict__ nnIdx) {
    int b = blockIdx.x;
    float best = -1.f; int bi = PPB; unsigned bnn = 0;
    for (int p = threadIdx.x; p < PPB; p += 256) {
        unsigned long long k = min64[b * PPB + p];
        float d2 = __uint_as_float((unsigned)(k >> 32));
        if (d2 > best) { best = d2; bi = p; bnn = (unsigned)(k & 0xFFFFFFFFu); }
    }
    __shared__ float sv[256]; __shared__ int si[256]; __shared__ unsigned sn[256];
    sv[threadIdx.x] = best; si[threadIdx.x] = bi; sn[threadIdx.x] = bnn;
    __syncthreads();
    for (int st = 128; st > 0; st >>= 1) {
        if (threadIdx.x < st) {
            float ov = sv[threadIdx.x + st]; int oi = si[threadIdx.x + st];
            if (ov > sv[threadIdx.x] || (ov == sv[threadIdx.x] && oi < si[threadIdx.x])) {
                sv[threadIdx.x] = ov; si[threadIdx.x] = oi; sn[threadIdx.x] = sn[threadIdx.x + st];
            }
        }
        __syncthreads();
    }
    if (threadIdx.x == 0) {
        maxPatch[b] = si[0];
        score[b]    = sqrtf(fmaxf(sv[0], 0.f));
        nnIdx[b]    = (int)sn[0];
    }
}

// ---- kernel 4: stage-2 distances, 16 x 16384 GEMM via WMMA ----------------

__global__ __launch_bounds__(256)
void dist2_kernel(const float* __restrict__ mb, const float* __restrict__ y2,
                  const int* __restrict__ nnIdx, float* __restrict__ s2d2) {
    __shared__ unsigned short Ahi[16 * LDT];
    __shared__ unsigned short Alo[16 * LDT];
    __shared__ unsigned short Bhi[128 * LDT];
    __shared__ unsigned short Blo[128 * LDT];
    __shared__ int rowsS[16];

    const int tid  = threadIdx.x;
    const int lane = tid & 31;
    const int wv   = tid >> 5;
    const int lm   = lane & 15;
    const int h    = lane >> 4;
    const int c0   = blockIdx.x * 128;

    if (tid < 16) rowsS[tid] = nnIdx[tid];
    __syncthreads();

    v8f acc = {0.f,0.f,0.f,0.f,0.f,0.f,0.f,0.f};

    for (int kk = 0; kk < DIM; kk += KC) {
        __syncthreads();
        if (tid < 128) {
            int flat = tid * 4;
            int rowi = flat >> 5;
            int ki   = flat & 31;
            v4f av = *(const v4f*)(mb + (size_t)rowsS[rowi] * DIM + kk + ki);
            v4u16 ah, al;
            #pragma unroll
            for (int j = 0; j < 4; ++j) {
                unsigned short hh = bf16_rne(av[j]);
                ah[j] = hh;
                al[j] = bf16_rne(av[j] - __uint_as_float((unsigned)hh << 16));
            }
            int off = rowi * LDT + ki;
            *(v4u16*)(Ahi + off) = ah;
            *(v4u16*)(Alo + off) = al;
        }
        #pragma unroll
        for (int i = 0; i < 4; ++i) {
            int idx  = tid + 256 * i;
            int flat = idx * 4;
            int rowi = flat >> 5;
            int ki   = flat & 31;
            v4f bv = *(const v4f*)(mb + (size_t)(c0 + rowi) * DIM + kk + ki);
            v4u16 bh, bl;
            #pragma unroll
            for (int j = 0; j < 4; ++j) {
                unsigned short hh = bf16_rne(bv[j]);
                bh[j] = hh;
                bl[j] = bf16_rne(bv[j] - __uint_as_float((unsigned)hh << 16));
            }
            int off = rowi * LDT + ki;
            *(v4u16*)(Bhi + off) = bh;
            *(v4u16*)(Blo + off) = bl;
        }
        __syncthreads();

        const unsigned short* pa = Ahi + lm * LDT + h * 8;
        const unsigned short* pl = Alo + lm * LDT + h * 8;
        v16bf a_hi = mk16(pa, pa + 16);
        v16bf a_lo = mk16(pl, pl + 16);
        const unsigned short* pbh = Bhi + (wv * 16 + lm) * LDT + h * 8;
        const unsigned short* pbl = Blo + (wv * 16 + lm) * LDT + h * 8;
        v16bf b_hi = mk16(pbh, pbh + 16);
        v16bf b_lo = mk16(pbl, pbl + 16);
        acc = WMMA_BF16(a_hi, b_hi, acc);
        acc = WMMA_BF16(a_hi, b_lo, acc);
        acc = WMMA_BF16(a_lo, b_hi, acc);
    }

    int col = c0 + wv * 16 + lm;
    float yc = y2[col];
    #pragma unroll
    for (int r = 0; r < 8; ++r) {
        int bimg = r + 8 * h;
        float d2 = fmaxf(y2[rowsS[bimg]] + yc - 2.0f * acc[r], 0.0f);
        s2d2[(size_t)bimg * M_MB + col] = d2;
    }
}

// ---- kernel 5: top-9 selection, support distances, softmax weight ---------

__global__ __launch_bounds__(256)
void final_kernel(const float* __restrict__ emb, const float* __restrict__ mb,
                  float* __restrict__ s2d2, const int* __restrict__ maxPatch,
                  const float* __restrict__ score, float* __restrict__ out) {
    int b = blockIdx.x;
    float* row = s2d2 + (size_t)b * M_MB;
    __shared__ float sv[256]; __shared__ int si[256];
    __shared__ int   support[9];
    __shared__ float dists[9];

    for (int j = 0; j < 9; ++j) {
        float best = 3.0e38f; int bi = M_MB;
        for (int m = threadIdx.x; m < M_MB; m += 256) {
            float v = row[m];
            if (v < best) { best = v; bi = m; }
        }
        sv[threadIdx.x] = best; si[threadIdx.x] = bi;
        __syncthreads();
        for (int st = 128; st > 0; st >>= 1) {
            if (threadIdx.x < st) {
                float ov = sv[threadIdx.x + st]; int oi = si[threadIdx.x + st];
                if (ov < sv[threadIdx.x] || (ov == sv[threadIdx.x] && oi < si[threadIdx.x])) {
                    sv[threadIdx.x] = ov; si[threadIdx.x] = oi;
                }
            }
            __syncthreads();
        }
        if (threadIdx.x == 0) { support[j] = si[0]; row[si[0]] = 3.0e38f; }
        __syncthreads();
    }

    const float* feat = emb + (size_t)(b * PPB + maxPatch[b]) * DIM;
    for (int j = 0; j < 9; ++j) {
        const float* sup = mb + (size_t)support[j] * DIM;
        float s = 0.f;
        for (int k = threadIdx.x; k < DIM; k += 256) {
            float df = feat[k] - sup[k];
            s += df * df;
        }
        for (int m = 16; m >= 1; m >>= 1) s += __shfl_xor(s, m);
        if ((threadIdx.x & 31) == 0) sv[threadIdx.x >> 5] = s;
        __syncthreads();
        if (threadIdx.x == 0) {
            float t = 0.f;
            for (int i = 0; i < 8; ++i) t += sv[i];
            dists[j] = sqrtf(fmaxf(t, 0.f));
        }
        __syncthreads();
    }

    if (threadIdx.x == 0) {
        float mx = dists[0];
        for (int j = 1; j < 9; ++j) mx = fmaxf(mx, dists[j]);
        float sum = 0.f, e0 = 0.f;
        for (int j = 0; j < 9; ++j) {
            float e = expf(dists[j] - mx);
            sum += e;
            if (j == 0) e0 = e;
        }
        out[b] = (1.f - e0 / sum) * score[b];
    }
}

// ---- host launch -----------------------------------------------------------

extern "C" void kernel_launch(void* const* d_in, const int* in_sizes, int n_in,
                              void* d_out, int out_size, void* d_ws, size_t ws_size,
                              hipStream_t stream) {
    const float* emb = (const float*)d_in[0];   // [12544,1536] f32
    const float* mb  = (const float*)d_in[1];   // [16384,1536] f32
    (void)in_sizes; (void)n_in; (void)out_size;

    char* ws = (char*)d_ws;
    float*              x2       = (float*)(ws + OFF_X2);
    float*              y2       = (float*)(ws + OFF_Y2);
    unsigned long long* min64    = (unsigned long long*)(ws + OFF_MIN64);
    int*                maxPatch = (int*)(ws + OFF_MAXP);
    float*              score    = (float*)(ws + OFF_SCORE);
    int*                nnIdx    = (int*)(ws + OFF_NNIDX);
    float*              s2d2     = (float*)(ws + OFF_S2D2);
    unsigned short*     ehi      = (unsigned short*)(ws + OFF_EHI);
    unsigned short*     elo      = (unsigned short*)(ws + OFF_ELO);
    unsigned short*     mhi      = (unsigned short*)(ws + OFF_MHI);
    unsigned short*     mlo      = (unsigned short*)(ws + OFF_MLO);
    float*              out      = (float*)d_out;

    const int fast = (ws_size >= WS_REQ_FAST) ? 1 : 0;

    prep_kernel<<<N_EMB + M_MB, 256, 0, stream>>>(emb, mb, ehi, elo, mhi, mlo,
                                                  x2, y2, min64, fast);
    dim3 g2(98, NSPLIT);
    if (fast) {
        dist_argmin_fast<<<g2, 256, 0, stream>>>(ehi, elo, mhi, mlo, x2, y2, min64);
    } else {
        dist_argmin_kernel<<<g2, 256, 0, stream>>>(emb, mb, x2, y2, min64);
    }
    argmax_kernel<<<BATCH, 256, 0, stream>>>(min64, maxPatch, score, nnIdx);
    dist2_kernel<<<M_MB / 128, 256, 0, stream>>>(mb, y2, nnIdx, s2d2);
    final_kernel<<<BATCH, 256, 0, stream>>>(emb, mb, s2d2, maxPatch, score, out);
}
